// CausalMultiHeadAttention_43911745634979
// MI455X (gfx1250) — matrix operand order repaired and checked by reading, not compiled, not run on hardware
//
#include <hip/hip_runtime.h>
#include <hip/hip_bf16.h>

#define BATCH 4
#define SEQ   2048
#define EMB   1024
#define NH    16
#define HD    64
#define MTOT  (BATCH*SEQ)   // 8192

typedef __attribute__((ext_vector_type(16))) __bf16 v16bf;
typedef __attribute__((ext_vector_type(8)))  __bf16 v8bf;
typedef __attribute__((ext_vector_type(8)))  float  v8f;
typedef __attribute__((ext_vector_type(4)))  unsigned int u32x4;
typedef __attribute__((ext_vector_type(4)))  int i32x4;
typedef __attribute__((ext_vector_type(8)))  int i32x8;

#if __has_builtin(__builtin_amdgcn_tensor_load_to_lds) && \
    __has_builtin(__builtin_amdgcn_s_wait_tensorcnt)
#define USE_TDM 1
#else
#define USE_TDM 0
#endif

static __device__ __forceinline__ v8f wmma_bf16(v16bf a, v16bf b, v8f c) {
  return __builtin_amdgcn_wmma_f32_16x16x32_bf16(false, a, false, b, (short)0, c, false, false);
}

static __device__ __forceinline__ v16bf cat16(v8bf lo, v8bf hi) {
  v16bf r;
#pragma unroll
  for (int i = 0; i < 8; ++i) { r[i] = lo[i]; r[i + 8] = hi[i]; }
  return r;
}

static __device__ __forceinline__ v8bf cvt8(const float* __restrict__ p) {
  float4 a = *(const float4*)p;
  float4 b = *(const float4*)(p + 4);
  v8bf r;
  r[0] = (__bf16)a.x; r[1] = (__bf16)a.y; r[2] = (__bf16)a.z; r[3] = (__bf16)a.w;
  r[4] = (__bf16)b.x; r[5] = (__bf16)b.y; r[6] = (__bf16)b.z; r[7] = (__bf16)b.w;
  return r;
}

#define LDSS 40  // padded LDS row stride in bf16 (80B rows: 64B data + 16B pad)

// ---------------------------------------------------------------------------
// TDM: DMA one 128x32 bf16 tile (row stride EMB elems) into LDS with the 16B
// row padding applied by the TDM pad unit (pad_interval=16DW, pad_amount=4DW).
// ---------------------------------------------------------------------------
#if USE_TDM
static __device__ __forceinline__ void tdm_load_tile(const __bf16* gsrc,
                                                     unsigned lds_byte_off) {
  unsigned long long ga = (unsigned long long)(uintptr_t)gsrc;
  u32x4 g0;
  g0[0] = 1u;                                   // count=1, user descriptor
  g0[1] = lds_byte_off;                          // lds_addr
  g0[2] = (unsigned)(ga & 0xffffffffu);          // global_addr[31:0]
  g0[3] = (unsigned)((ga >> 32) & 0x01ffffffu)   // global_addr[56:32]
        | 0x80000000u;                           // type=2 ("image")
  i32x8 g1;
  g1[0] = (1 << 16)      // data_size = 2B
        | (1 << 20)      // pad_enable
        | (3 << 22)      // pad_interval: 16 DWORDs (64B) between pads
        | (3 << 25);     // pad_amount: 4 DWORDs (16B)
  g1[1] = (32 << 16);                            // tensor_dim0 = 32 (lo16<<16)
  g1[2] = (128 << 16);                           // dim0 hi=0 | tensor_dim1=128
  g1[3] = (32 << 16);                            // dim1 hi=0 | tile_dim0=32
  g1[4] = 128;                                   // tile_dim1 = 128
  g1[5] = EMB;                                   // tensor_dim0_stride = 1024
  g1[6] = 0;
  g1[7] = 0;
  i32x4 z4 = {0, 0, 0, 0};
#if __clang_major__ >= 23
  i32x8 z8 = {0, 0, 0, 0, 0, 0, 0, 0};
  __builtin_amdgcn_tensor_load_to_lds(g0, g1, z4, z4, z8, 0);
#else
  __builtin_amdgcn_tensor_load_to_lds(g0, g1, z4, z4, 0);
#endif
}
#endif

// stage one 128x32 bf16 tile (row stride EMB) into padded LDS
static __device__ __forceinline__ void stage_tile(__bf16* ldsbase,
                                                  const __bf16* src, int t) {
#if USE_TDM
  if (t < 32) tdm_load_tile(src, (unsigned)(uintptr_t)ldsbase);
#else
  const int r = t >> 1, half = (t & 1) * 16;
  const __bf16* s = src + (size_t)r * EMB + half;
  __bf16* d = ldsbase + r * LDSS + half;
  *(v8bf*)d       = *(const v8bf*)s;
  *(v8bf*)(d + 8) = *(const v8bf*)(s + 8);
#endif
}

static __device__ __forceinline__ void wait_stage(int t, bool has_next) {
#if USE_TDM
  if (t < 32) {
    if (has_next) __builtin_amdgcn_s_wait_tensorcnt(2);  // current tile done
    else          __builtin_amdgcn_s_wait_tensorcnt(0);
  }
#endif
}

// ---------------------------------------------------------------------------
// Pre-pass 1: x (fp32) -> xb (bf16), elementwise. grid = 4096 x 256.
// ---------------------------------------------------------------------------
__global__ __launch_bounds__(256) void cvt_x(const float* __restrict__ in,
                                             __bf16* __restrict__ out) {
  const size_t i = ((size_t)blockIdx.x * 256 + threadIdx.x) * 8;
  *(v8bf*)(out + i) = cvt8(in + i);
}

// ---------------------------------------------------------------------------
// Pre-pass 2: W[k][n] fp32 -> Wt[n][k] bf16 (32x32 LDS tile transpose).
// grid = (32, 32, 4), block = 256. z selects Wq/Wk/Wv/Wo.
// ---------------------------------------------------------------------------
__global__ __launch_bounds__(256) void cvt_w(
    const float* __restrict__ Wq, const float* __restrict__ Wk,
    const float* __restrict__ Wv, const float* __restrict__ Wo,
    __bf16* __restrict__ Wt) {
  __shared__ __bf16 tile[32][33];
  const float* W = (blockIdx.z == 0) ? Wq : (blockIdx.z == 1) ? Wk
                 : (blockIdx.z == 2) ? Wv : Wo;
  __bf16* Out = Wt + (size_t)blockIdx.z * EMB * EMB;
  const int c = threadIdx.x & 31, r = threadIdx.x >> 5;
  const int k0 = blockIdx.x * 32, n0 = blockIdx.y * 32;
#pragma unroll
  for (int i = 0; i < 4; ++i) {
    const int kk = r * 4 + i;
    tile[c][kk] = (__bf16)W[(size_t)(k0 + kk) * EMB + n0 + c];  // coalesced read
  }
  __syncthreads();
#pragma unroll
  for (int i = 0; i < 4; ++i) {
    const int nn = r * 4 + i;
    Out[(size_t)(n0 + nn) * EMB + k0 + c] = tile[nn][c];        // coalesced write
  }
}

// ---------------------------------------------------------------------------
// Kernel 3: QKV GEMM, all-bf16 tiles, TDM double-buffered staging.
// Q, K written as [B,H,S,D]; V written TRANSPOSED as [B,H,D,S] so that the
// attention P*V B-fragments become contiguous loads.
// grid = (EMB/128, MTOT/128, 3), block = 256 (8 waves).
// ---------------------------------------------------------------------------
__global__ __launch_bounds__(256) void qkv_gemm(
    const __bf16* __restrict__ xb, const __bf16* __restrict__ Wt,
    __bf16* __restrict__ Qo, __bf16* __restrict__ Ko, __bf16* __restrict__ Vt) {
  __shared__ __align__(16) __bf16 Alds[2][128 * LDSS];
  __shared__ __align__(16) __bf16 Blds[2][128 * LDSS];

  const int t    = threadIdx.x;
  const int lane = t & 31;
  const int wid  = t >> 5;
  const int wm   = wid & 3;
  const int wn   = wid >> 2;
  const int r    = lane & 15;
  const int hi   = lane >> 4;

  const int n0 = blockIdx.x * 128;
  const int m0 = blockIdx.y * 128;

  const __bf16* Abase = xb + (size_t)m0 * EMB;
  const __bf16* Bbase = Wt + (size_t)blockIdx.z * EMB * EMB + (size_t)n0 * EMB;

  v8f acc[2][4];
#pragma unroll
  for (int i = 0; i < 2; ++i)
#pragma unroll
    for (int j = 0; j < 4; ++j)
      acc[i][j] = v8f{0.f,0.f,0.f,0.f,0.f,0.f,0.f,0.f};

  stage_tile(&Alds[0][0], Abase, t);
  stage_tile(&Blds[0][0], Bbase, t);

  for (int kc = 0; kc < EMB; kc += 32) {
    const int cur = (kc >> 5) & 1;
    const bool has_next = (kc + 32) < EMB;
    if (has_next) {  // prefetch next tile into the other buffer
      stage_tile(&Alds[cur ^ 1][0], Abase + kc + 32, t);
      stage_tile(&Blds[cur ^ 1][0], Bbase + kc + 32, t);
    }
    wait_stage(t, has_next);
    __syncthreads();

    v16bf af[2];
#pragma unroll
    for (int i = 0; i < 2; ++i) {
      const __bf16* p = &Alds[cur][(wm * 32 + i * 16 + r) * LDSS];
      af[i] = cat16(*(const v8bf*)(p + hi * 16), *(const v8bf*)(p + hi * 16 + 8));
    }
#pragma unroll
    for (int j = 0; j < 4; ++j) {
      const __bf16* p = &Blds[cur][(wn * 64 + j * 16 + r) * LDSS + hi * 16];
      v16bf bfg = cat16(*(const v8bf*)p, *(const v8bf*)(p + 8));
#pragma unroll
      for (int i = 0; i < 2; ++i) acc[i][j] = wmma_bf16(af[i], bfg, acc[i][j]);
    }
    __syncthreads();
  }

  const int z = blockIdx.z;
#pragma unroll
  for (int i = 0; i < 2; ++i) {
#pragma unroll
    for (int j = 0; j < 4; ++j) {
      const int n = n0 + wn * 64 + j * 16 + r;
      const int h = n >> 6, d = n & 63;
#pragma unroll
      for (int e = 0; e < 8; ++e) {
        const int m = m0 + wm * 32 + i * 16 + e + 8 * hi;
        const int b = m >> 11, s = m & (SEQ - 1);
        const __bf16 val = (__bf16)acc[i][j][e];
        if (z == 0) {
          Qo[((size_t)(b * NH + h) * SEQ + s) * HD + d] = val;        // [B,H,S,D]
        } else if (z == 1) {
          Ko[((size_t)(b * NH + h) * SEQ + s) * HD + d] = val;        // [B,H,S,D]
        } else {
          Vt[((size_t)(b * NH + h) * HD + d) * SEQ + s] = val;        // [B,H,D,S]
        }
      }
    }
  }
}

// ---------------------------------------------------------------------------
// Kernel 4: causal flash attention. grid = (SEQ/64, B*H), block = 128.
// V is in transposed [B,H,D,S] layout: P*V B-frags are contiguous loads.
// ---------------------------------------------------------------------------
__global__ __launch_bounds__(128) void attn(
    const __bf16* __restrict__ Q, const __bf16* __restrict__ K,
    const __bf16* __restrict__ Vt, __bf16* __restrict__ O) {
  __shared__ __align__(16) __bf16 Plds[4][16][LDSS];

  const int t = threadIdx.x;
  const int lane = t & 31;
  const int wid = t >> 5;
  const int r = lane & 15;
  const int hi = lane >> 4;

  const int bh = blockIdx.y;
  const int b = bh >> 4, h = bh & 15;
  const int qbase = blockIdx.x * 64 + wid * 16;
  const int qend = qbase + 15;

  const __bf16* Qb = Q  + (size_t)bh * SEQ * HD;
  const __bf16* Kb = K  + (size_t)bh * SEQ * HD;
  const __bf16* Vb = Vt + (size_t)bh * HD * SEQ;   // [d][s]

  v16bf qa[2];
#pragma unroll
  for (int dc = 0; dc < 2; ++dc) {
    const __bf16* p = Qb + (size_t)(qbase + r) * HD + dc * 32 + hi * 16;
    qa[dc] = cat16(*(const v8bf*)p, *(const v8bf*)(p + 8));
  }

  v8f o[4];
#pragma unroll
  for (int j = 0; j < 4; ++j) o[j] = v8f{0.f,0.f,0.f,0.f,0.f,0.f,0.f,0.f};
  float mrow[8], lrow[8];
#pragma unroll
  for (int e = 0; e < 8; ++e) { mrow[e] = -__builtin_inff(); lrow[e] = 0.f; }

  const float scale = 0.125f;  // 1/sqrt(64)

  for (int k0 = 0; k0 <= qend; k0 += 32) {
    // --- S = Q K^T for 16 queries x 32 keys ---
    v8f s0 = {0.f,0.f,0.f,0.f,0.f,0.f,0.f,0.f};
    v8f s1 = {0.f,0.f,0.f,0.f,0.f,0.f,0.f,0.f};
#pragma unroll
    for (int dc = 0; dc < 2; ++dc) {
      const __bf16* kp0 = Kb + (size_t)(k0 + r) * HD + dc * 32 + hi * 16;
      v16bf kb0 = cat16(*(const v8bf*)kp0, *(const v8bf*)(kp0 + 8));
      const __bf16* kp1 = Kb + (size_t)(k0 + 16 + r) * HD + dc * 32 + hi * 16;
      v16bf kb1 = cat16(*(const v8bf*)kp1, *(const v8bf*)(kp1 + 8));
      s0 = wmma_bf16(qa[dc], kb0, s0);
      s1 = wmma_bf16(qa[dc], kb1, s1);
    }

    // --- causal mask + scale + online softmax ---
    float alpha[8];
#pragma unroll
    for (int e = 0; e < 8; ++e) {
      const int q = qbase + e + 8 * hi;
      const int c0 = k0 + r, c1 = c0 + 16;
      float v0 = (c0 <= q) ? s0[e] * scale : -__builtin_inff();
      float v1 = (c1 <= q) ? s1[e] * scale : -__builtin_inff();
      float mx = fmaxf(v0, v1);
      mx = fmaxf(mx, __shfl_xor(mx, 1, 16));
      mx = fmaxf(mx, __shfl_xor(mx, 2, 16));
      mx = fmaxf(mx, __shfl_xor(mx, 4, 16));
      mx = fmaxf(mx, __shfl_xor(mx, 8, 16));
      const float mnew = fmaxf(mrow[e], mx);
      const float e0 = __expf(v0 - mnew);
      const float e1 = __expf(v1 - mnew);
      float sum = e0 + e1;
      sum += __shfl_xor(sum, 1, 16);
      sum += __shfl_xor(sum, 2, 16);
      sum += __shfl_xor(sum, 4, 16);
      sum += __shfl_xor(sum, 8, 16);
      alpha[e] = __expf(mrow[e] - mnew);
      lrow[e] = lrow[e] * alpha[e] + sum;
      mrow[e] = mnew;
      Plds[wid][e + 8 * hi][r]      = (__bf16)e0;
      Plds[wid][e + 8 * hi][r + 16] = (__bf16)e1;
    }

#pragma unroll
    for (int j = 0; j < 4; ++j)
#pragma unroll
      for (int e = 0; e < 8; ++e) o[j][e] *= alpha[e];

    asm volatile("s_wait_dscnt 0" ::: "memory");  // wave-local LDS W->R order

    const __bf16* pp = &Plds[wid][r][0];
    v16bf pa = cat16(*(const v8bf*)(pp + hi * 16), *(const v8bf*)(pp + hi * 16 + 8));

    // --- O += P V : B-frag lane = column d, 16 consecutive keys (contiguous) ---
#pragma unroll
    for (int j = 0; j < 4; ++j) {
      const __bf16* vp = Vb + (size_t)(j * 16 + r) * SEQ + k0 + hi * 16;
      v16bf vbf = cat16(*(const v8bf*)vp, *(const v8bf*)(vp + 8));
      o[j] = wmma_bf16(pa, vbf, o[j]);
    }
  }

  float rinv[8];
#pragma unroll
  for (int e = 0; e < 8; ++e) rinv[e] = 1.0f / lrow[e];
#pragma unroll
  for (int j = 0; j < 4; ++j) {
#pragma unroll
    for (int e = 0; e < 8; ++e) {
      const int q = qbase + e + 8 * hi;
      const int d = j * 16 + r;
      O[(size_t)(b * SEQ + q) * EMB + h * HD + d] = (__bf16)(o[j][e] * rinv[e]);
    }
  }
}

// ---------------------------------------------------------------------------
// Kernel 5: output projection y = A @ Wo + bo (bf16 in, fp32 out).
// grid = (EMB/128, MTOT/128), block = 256.
// ---------------------------------------------------------------------------
__global__ __launch_bounds__(256) void out_gemm(
    const __bf16* __restrict__ A, const __bf16* __restrict__ Wot,
    const float* __restrict__ bo, float* __restrict__ Y) {
  __shared__ __align__(16) __bf16 Alds[2][128 * LDSS];
  __shared__ __align__(16) __bf16 Blds[2][128 * LDSS];

  const int t    = threadIdx.x;
  const int lane = t & 31;
  const int wid  = t >> 5;
  const int wm   = wid & 3;
  const int wn   = wid >> 2;
  const int r    = lane & 15;
  const int hi   = lane >> 4;

  const int n0 = blockIdx.x * 128;
  const int m0 = blockIdx.y * 128;

  const __bf16* Abase = A + (size_t)m0 * EMB;
  const __bf16* Bbase = Wot + (size_t)n0 * EMB;

  v8f acc[2][4];
#pragma unroll
  for (int i = 0; i < 2; ++i)
#pragma unroll
    for (int j = 0; j < 4; ++j)
      acc[i][j] = v8f{0.f,0.f,0.f,0.f,0.f,0.f,0.f,0.f};

  stage_tile(&Alds[0][0], Abase, t);
  stage_tile(&Blds[0][0], Bbase, t);

  for (int kc = 0; kc < EMB; kc += 32) {
    const int cur = (kc >> 5) & 1;
    const bool has_next = (kc + 32) < EMB;
    if (has_next) {
      stage_tile(&Alds[cur ^ 1][0], Abase + kc + 32, t);
      stage_tile(&Blds[cur ^ 1][0], Bbase + kc + 32, t);
    }
    wait_stage(t, has_next);
    __syncthreads();

    v16bf af[2];
#pragma unroll
    for (int i = 0; i < 2; ++i) {
      const __bf16* p = &Alds[cur][(wm * 32 + i * 16 + r) * LDSS];
      af[i] = cat16(*(const v8bf*)(p + hi * 16), *(const v8bf*)(p + hi * 16 + 8));
    }
#pragma unroll
    for (int j = 0; j < 4; ++j) {
      const __bf16* p = &Blds[cur][(wn * 64 + j * 16 + r) * LDSS + hi * 16];
      v16bf bfg = cat16(*(const v8bf*)p, *(const v8bf*)(p + 8));
#pragma unroll
      for (int i = 0; i < 2; ++i) acc[i][j] = wmma_bf16(af[i], bfg, acc[i][j]);
    }
    __syncthreads();
  }

#pragma unroll
  for (int j = 0; j < 4; ++j) {
    const int n = n0 + wn * 64 + j * 16 + r;
    const float bias = bo[n];
#pragma unroll
    for (int i = 0; i < 2; ++i) {
#pragma unroll
      for (int e = 0; e < 8; ++e) {
        const int m = m0 + wm * 32 + i * 16 + e + 8 * hi;
        Y[(size_t)m * EMB + n] = acc[i][j][e] + bias;
      }
    }
  }
}

// ---------------------------------------------------------------------------
extern "C" void kernel_launch(void* const* d_in, const int* in_sizes, int n_in,
                              void* d_out, int out_size, void* d_ws, size_t ws_size,
                              hipStream_t stream) {
  const float* x  = (const float*)d_in[0];
  const float* Wq = (const float*)d_in[1];
  const float* Wk = (const float*)d_in[2];
  const float* Wv = (const float*)d_in[3];
  const float* Wo = (const float*)d_in[4];
  const float* bo = (const float*)d_in[5];
  float* y = (float*)d_out;

  // workspace (bf16 elems), ~76 MB total:
  //   [0, NQ)        xb  (x in bf16)  -- later reused as attention output Ab
  //   [NQ, 4*NQ)     Q, K in [B,H,S,D]; V in [B,H,D,S]
  //   [4*NQ, +4*E*E) Wt: 4 transposed bf16 weights [n][k]
  __bf16* ws = (__bf16*)d_ws;
  const size_t NQ = (size_t)BATCH * NH * SEQ * HD;  // 8,388,608
  __bf16* xb = ws;
  __bf16* Qb = ws + NQ;
  __bf16* Kb = ws + 2 * NQ;
  __bf16* Vb = ws + 3 * NQ;
  __bf16* Wt = ws + 4 * NQ;
  __bf16* Ab = xb;  // alias: xb dead after qkv_gemm

  cvt_x<<<dim3((MTOT * (size_t)EMB) / 2048), 256, 0, stream>>>(x, xb);
  cvt_w<<<dim3(EMB / 32, EMB / 32, 4), 256, 0, stream>>>(Wq, Wk, Wv, Wo, Wt);
  qkv_gemm<<<dim3(EMB / 128, MTOT / 128, 3), 256, 0, stream>>>(
      xb, Wt, Qb, Kb, Vb);
  attn<<<dim3(SEQ / 64, BATCH * NH), 128, 0, stream>>>(Qb, Kb, Vb, Ab);
  out_gemm<<<dim3(EMB / 128, MTOT / 128), 256, 0, stream>>>(
      Ab, Wt + 3 * (size_t)EMB * EMB, bo, y);
}
